// LshSelfAttention_41893111005512
// MI455X (gfx1250) — matrix operand order repaired and checked by reading, not compiled, not run on hardware
//
#include <hip/hip_runtime.h>
#include <hip/hip_bf16.h>

typedef __attribute__((ext_vector_type(16))) _Float16 v16h;
typedef __attribute__((ext_vector_type(8)))  _Float16 v8h;
typedef __attribute__((ext_vector_type(8)))  float    v8f;
typedef __attribute__((ext_vector_type(4)))  unsigned int v4u;
typedef __attribute__((ext_vector_type(8)))  int v8i;
typedef __attribute__((ext_vector_type(4)))  int v4i;

namespace {
constexpr int kB       = 4;
constexpr int kL       = 4096;
constexpr int kHid     = 512;
constexpr int kH       = 8;
constexpr int kE       = 64;     // head dim
constexpr int kNH      = 2;      // hash rounds
constexpr int kBucket  = 64;
constexpr int kNBuck   = 64;     // L / bucket
constexpr int kBH      = kB * kH;          // 32
constexpr int kSN      = kNH * kL;         // 8192 sorted rows per BH
constexpr int kNCh     = kSN / kBucket;    // 128 chunks per BH
constexpr int kNKeys   = kNH * kNBuck;     // 128 distinct sort buckets
}

// ---------------- WMMA fragment loaders (wave32, 16x16x32 f16) ----------------
__device__ __forceinline__ v16h load_a_frag(const _Float16* p, int ld) {
  int lane = threadIdx.x & 31;
  int m  = lane & 15;
  int kb = (lane >> 4) * 16;
  const _Float16* r = p + m * ld + kb;
  v8h lo = *(const v8h*)(r);
  v8h hi = *(const v8h*)(r + 8);
  v16h a;
#pragma unroll
  for (int i = 0; i < 8; ++i) { a[i] = lo[i]; a[i + 8] = hi[i]; }
  return a;
}

__device__ __forceinline__ v16h load_b_frag(const _Float16* p, int ld) {
  int lane = threadIdx.x & 31;
  int n  = lane & 15;
  int kb = (lane >> 4) * 16;
  return *(const v16h*)(p + n * ld + kb);
}

// ---------------- Tensor Data Mover: 1-D contiguous f16 copy to LDS ----------------
__device__ __forceinline__ void tdm_load_1d_f16(void* lds_ptr, const void* gsrc,
                                                unsigned nelem) {
  unsigned lds_addr = (unsigned)(unsigned long long)lds_ptr;   // low 32 bits = LDS offset
  unsigned long long ga = (unsigned long long)gsrc;
  v4u g0;
  g0[0] = 1u;                                            // count=1, user descriptor
  g0[1] = lds_addr;                                      // lds_addr
  g0[2] = (unsigned)(ga & 0xFFFFFFFFu);                  // global_addr[31:0]
  g0[3] = (unsigned)((ga >> 32) & 0x1FFFFFFu) | (2u << 30);  // global_addr[56:32] | type=2
  v8i g1;
  g1[0] = 1 << 16;                                       // data_size = 1 (2 bytes)
  g1[1] = (int)((nelem & 0xFFFFu) << 16);                // tensor_dim0[15:0]
  g1[2] = (int)((nelem >> 16) | (1u << 16));             // tensor_dim0[31:16] | tensor_dim1=1
  g1[3] = (int)(nelem << 16);                            // tile_dim0 = nelem (<=65535)
  g1[4] = 0;                                             // tile_dim1=0, tile_dim2=0 (1-D)
  g1[5] = (int)nelem;                                    // tensor_dim0_stride lo
  g1[6] = 0;
  g1[7] = 0;
  v4i gz = {0, 0, 0, 0};
#if defined(__clang_major__) && __clang_major__ >= 23
  v8i gz8 = {0, 0, 0, 0, 0, 0, 0, 0};
  __builtin_amdgcn_tensor_load_to_lds(g0, g1, gz, gz, gz8, 0);
#else
  __builtin_amdgcn_tensor_load_to_lds(g0, g1, gz, gz, 0);
#endif
}

// ---------------- elementwise converts ----------------
__global__ void conv_f32_to_f16(const float* __restrict__ x, _Float16* __restrict__ y, int n) {
  int i = blockIdx.x * blockDim.x + threadIdx.x;
  if (i < n) y[i] = (_Float16)x[i];
}

// src: row-major [K=512][N=512] f32 -> dst: row-major [N][K] f16 (transposed)
__global__ void conv_transpose_w(const float* __restrict__ w, _Float16* __restrict__ wt) {
  int i = blockIdx.x * blockDim.x + threadIdx.x;
  if (i >= kHid * kHid) return;
  int k = i / kHid, n = i % kHid;
  wt[(size_t)n * kHid + k] = (_Float16)w[i];
}

// ---------------- WMMA GEMM with double-buffered LDS staging ----------------
// C[M,N] = A[M,K](f16) * Bt[N,K]^T(f16); block = 128 threads, 64x64 C tile, K step 64.
// Fragment loads are batched per macro-step so one coarse dscnt wait covers the
// whole 8-WMMA chain instead of one wait per WMMA.
__global__ void gemm_f16f32(const _Float16* __restrict__ A, const _Float16* __restrict__ Bt,
                            float* __restrict__ C, int M, int N, int K, int mode) {
  __shared__ alignas(32) _Float16 lA[2][64 * 64];
  __shared__ alignas(32) _Float16 lB[2][64 * 64];
  int wave = threadIdx.x >> 5;
  int lane = threadIdx.x & 31;
  int m0 = blockIdx.y * 64;
  int n0 = blockIdx.x * 64;
  const int KS = 64;
  int nsteps = K / KS;

  v8h ra[4], rb[4];
  auto load_stage = [&](int k0) {
#pragma unroll
    for (int j = 0; j < 4; ++j) {
      int c = (int)threadIdx.x + j * 128;          // chunk of 8 halves
      int row = c >> 3, col = (c & 7) * 8;
      ra[j] = *(const v8h*)(A + (size_t)(m0 + row) * K + k0 + col);
      rb[j] = *(const v8h*)(Bt + (size_t)(n0 + row) * K + k0 + col);
    }
  };
  auto store_stage = [&](int buf) {
#pragma unroll
    for (int j = 0; j < 4; ++j) {
      int c = (int)threadIdx.x + j * 128;
      *(v8h*)&lA[buf][c * 8] = ra[j];
      *(v8h*)&lB[buf][c * 8] = rb[j];
    }
  };

  v8f acc[4] = {};
  load_stage(0);
  store_stage(0);
  __syncthreads();
  for (int s = 0; s < nsteps; ++s) {
    if (s + 1 < nsteps) load_stage((s + 1) * KS);   // prefetch next macro-tile
    int buf = s & 1;
    // batch all fragment loads for this macro-step, then run the WMMA chain
    v16h a0 = load_a_frag(&lA[buf][(wave * 16) * 64 + 0], 64);
    v16h a1 = load_a_frag(&lA[buf][(wave * 16) * 64 + 32], 64);
    v16h bf[8];
#pragma unroll
    for (int t = 0; t < 4; ++t) {
      bf[t * 2]     = load_b_frag(&lB[buf][(t * 16) * 64 + 0], 64);
      bf[t * 2 + 1] = load_b_frag(&lB[buf][(t * 16) * 64 + 32], 64);
    }
#pragma unroll
    for (int t = 0; t < 4; ++t) {
      acc[t] = __builtin_amdgcn_wmma_f32_16x16x32_f16(false, a0, false, bf[t * 2],
                                                      (short)0, acc[t], false, false);
      acc[t] = __builtin_amdgcn_wmma_f32_16x16x32_f16(false, a1, false, bf[t * 2 + 1],
                                                      (short)0, acc[t], false, false);
    }
    __syncthreads();
    if (s + 1 < nsteps) store_stage(buf ^ 1);
    __syncthreads();
  }

  int mbase = m0 + wave * 16 + (lane >> 4) * 8;
  int nbase = n0 + (lane & 15);
#pragma unroll
  for (int t = 0; t < 4; ++t) {
#pragma unroll
    for (int r = 0; r < 8; ++r) {
      int m = mbase + r;
      int n = nbase + t * 16;
      float val = acc[t][r];
      if (mode == 0) {
        C[(size_t)m * N + n] = val;
      } else {
        int bb = m / kL, l = m % kL;
        int h = n >> 6, e = n & 63;
        C[(((size_t)(bb * kH + h)) * kL + l) * kE + e] = val;
      }
    }
  }
}

// ---------------- LSH hashing: argmax over [rot, -rot] ----------------
__global__ void hash_kernel(const float* __restrict__ qk, const float* __restrict__ rot,
                            int* __restrict__ buckets) {
  int idx = blockIdx.x * blockDim.x + threadIdx.x;
  if (idx >= kBH * kL) return;
  int bh = idx / kL, n = idx % kL;
  const float* q = qk + ((size_t)bh * kL + n) * kE;
  float qr[kE];
#pragma unroll
  for (int d = 0; d < kE; ++d) qr[d] = q[d];
  for (int h = 0; h < kNH; ++h) {
    float dots[kNBuck / 2];
#pragma unroll
    for (int j = 0; j < kNBuck / 2; ++j) {
      float acc = 0.f;
#pragma unroll
      for (int d = 0; d < kE; ++d)
        acc += qr[d] * rot[d * (kNH * (kNBuck / 2)) + h * (kNBuck / 2) + j];
      dots[j] = acc;
    }
    float best = -3.4e38f; int bestj = 0;
#pragma unroll
    for (int j = 0; j < kNBuck; ++j) {
      float v = (j < kNBuck / 2) ? dots[j] : -dots[j - kNBuck / 2];
      if (v > best) { best = v; bestj = j; }
    }
    buckets[(size_t)bh * kSN + h * kL + n] = bestj + h * kNBuck;
  }
}

// ---------------- stable counting sort per BH over 128 bucket ids ----------------
__global__ void sort_kernel(const int* __restrict__ buckets,
                            int* __restrict__ sticker, int* __restrict__ undo) {
  __shared__ int sb[kSN];
  __shared__ int counts[kNKeys];
  __shared__ int offs[kNKeys];
  int bh = blockIdx.x;
  for (int i = threadIdx.x; i < kSN; i += blockDim.x) sb[i] = buckets[(size_t)bh * kSN + i];
  if ((int)threadIdx.x < kNKeys) counts[threadIdx.x] = 0;
  __syncthreads();
  for (int i = threadIdx.x; i < kSN; i += blockDim.x) atomicAdd(&counts[sb[i]], 1);
  __syncthreads();
  if (threadIdx.x == 0) {
    int run = 0;
    for (int t = 0; t < kNKeys; ++t) { offs[t] = run; run += counts[t]; }
  }
  __syncthreads();
  int t = threadIdx.x;  // one thread owns one bucket -> stable, deterministic
  if (t < kNKeys) {
    int pos = offs[t];
    for (int i = 0; i < kSN; ++i) {
      if (sb[i] == t) {
        sticker[(size_t)bh * kSN + pos] = i;
        undo[(size_t)bh * kSN + i] = pos;
        pos++;
      }
    }
  }
}

// ---------------- gather sorted q / l2norm(k) / v, positions, key mask ----------------
__global__ void gather_kernel(const float* __restrict__ qk, const float* __restrict__ vv,
                              const unsigned char* __restrict__ pmask,
                              const int* __restrict__ sticker,
                              _Float16* __restrict__ sq, _Float16* __restrict__ sk,
                              _Float16* __restrict__ sv,
                              int* __restrict__ tpos, unsigned char* __restrict__ smask) {
  int idx = blockIdx.x * blockDim.x + threadIdx.x;
  if (idx >= kBH * kSN) return;
  int bh = idx / kSN;
  int i = sticker[idx];
  int pos = i & (kL - 1);
  tpos[idx] = pos;
  smask[idx] = pmask[(bh / kH) * kL + pos];
  const float* q = qk + ((size_t)bh * kL + pos) * kE;
  const float* w = vv + ((size_t)bh * kL + pos) * kE;
  float ss = 0.f;
#pragma unroll
  for (int d = 0; d < kE; ++d) { float x = q[d]; ss += x * x; }
  float rn = rsqrtf(fmaxf(ss, 1e-12f));
  _Float16* dq = sq + (size_t)idx * kE;
  _Float16* dk = sk + (size_t)idx * kE;
  _Float16* dv = sv + (size_t)idx * kE;
#pragma unroll
  for (int d = 0; d < kE; ++d) {
    float x = q[d];
    dq[d] = (_Float16)x;
    dk[d] = (_Float16)(x * rn);
    dv[d] = (_Float16)w[d];
  }
}

// ---------------- chunked local attention (one block per (chunk, BH)) ----------------
__global__ void attn_kernel(const _Float16* __restrict__ sq, const _Float16* __restrict__ sk,
                            const _Float16* __restrict__ sv,
                            const int* __restrict__ tpos, const unsigned char* __restrict__ smask,
                            float* __restrict__ bo, float* __restrict__ lse_s) {
  __shared__ alignas(32) char sbuf[24 * 1024];           // qs(8K)+ks(16K); reused as probs(16K)
  __shared__ alignas(32) _Float16 vT[kE * 2 * kBucket];  // 64 x 128, Bt layout for PV
  __shared__ alignas(32) float dotsbuf[kBucket * 2 * kBucket]; // 32K; head reused as vraw(16K)
  __shared__ int tq[kBucket];
  __shared__ int tk[2 * kBucket];
  __shared__ unsigned char km[2 * kBucket];

  _Float16* qs    = (_Float16*)sbuf;                 // 64 x 64
  _Float16* ksm   = (_Float16*)(sbuf + 8 * 1024);    // 128 x 64
  _Float16* probs = (_Float16*)sbuf;                 // 64 x 128 (after qs/ks dead)
  _Float16* vraw  = (_Float16*)dotsbuf;              // 128 x 64 (before dots live)
  float*    dots  = dotsbuf;                         // 64 x 128

  int c  = blockIdx.x;
  int bh = blockIdx.y;
  int cp = (c + kNCh - 1) & (kNCh - 1);
  size_t base  = ((size_t)bh * kSN + (size_t)c  * kBucket) * kE;
  size_t pbase = ((size_t)bh * kSN + (size_t)cp * kBucket) * kE;
  const unsigned kTile = kBucket * kE;               // 4096 f16 = 8KB contiguous

  // TDM async copies of the contiguous 8KB chunk tiles (wave 0 issues, EXEC-free op)
  if (threadIdx.x < 32) {
    tdm_load_1d_f16(qs,            sq + base,  kTile);
    tdm_load_1d_f16(ksm,           sk + base,  kTile);
    tdm_load_1d_f16(ksm + kTile,   sk + pbase, kTile);
    tdm_load_1d_f16(vraw,          sv + base,  kTile);
    tdm_load_1d_f16(vraw + kTile,  sv + pbase, kTile);
  }
  // meta loads overlap the TDM transfers
  for (int i = threadIdx.x; i < kBucket; i += blockDim.x)
    tq[i] = tpos[(size_t)bh * kSN + c * kBucket + i];
  for (int i = threadIdx.x; i < 2 * kBucket; i += blockDim.x) {
    int cc = (i < kBucket) ? c : cp;
    int o  = i & (kBucket - 1);
    tk[i] = tpos[(size_t)bh * kSN + cc * kBucket + o];
    km[i] = smask[(size_t)bh * kSN + cc * kBucket + o];
  }
  if (threadIdx.x < 32) __builtin_amdgcn_s_wait_tensorcnt(0);
  __syncthreads();

  // transpose v into Bt layout: vT[e][j] = v[j][e]
  for (int i = threadIdx.x; i < 2 * kBucket * kE; i += blockDim.x) {
    int j = i >> 6, d = i & 63;
    vT[d * 128 + j] = vraw[i];
  }
  __syncthreads();

  int wave = threadIdx.x >> 5;
  int lane = threadIdx.x & 31;
  const float scale = 0.125f;  // 64^-0.5

  // dots = (q . k^T) * scale + masks : 32 tiles, 4 per wave
#pragma unroll
  for (int t = 0; t < 4; ++t) {
    int tile = wave * 4 + t;
    int mi = tile >> 3, nj = tile & 7;
    // batch frag loads, then WMMA pair
    v16h a0 = load_a_frag(qs  + mi * 16 * kE + 0,  kE);
    v16h a1 = load_a_frag(qs  + mi * 16 * kE + 32, kE);
    v16h b0 = load_b_frag(ksm + nj * 16 * kE + 0,  kE);
    v16h b1 = load_b_frag(ksm + nj * 16 * kE + 32, kE);
    v8f acc = {};
    acc = __builtin_amdgcn_wmma_f32_16x16x32_f16(false, a0, false, b0,
                                                 (short)0, acc, false, false);
    acc = __builtin_amdgcn_wmma_f32_16x16x32_f16(false, a1, false, b1,
                                                 (short)0, acc, false, false);
    int mbase = mi * 16 + (lane >> 4) * 8;
    int ncol  = nj * 16 + (lane & 15);
#pragma unroll
    for (int r = 0; r < 8; ++r) {
      int m = mbase + r;
      float val = acc[r] * scale;
      if (km[ncol]) val = -1e9f;              // padding mask
      if (tq[m] == tk[ncol]) val = -1e5f;     // discourage self (applied after)
      dots[m * 128 + ncol] = val;
    }
  }
  __syncthreads();

  // row softmax + lse ; write f16 probs into reused qs/ks region
  if ((int)threadIdx.x < kBucket) {
    int m = threadIdx.x;
    float mx = -3.4e38f;
#pragma unroll 4
    for (int n = 0; n < 128; ++n) mx = fmaxf(mx, dots[m * 128 + n]);
    float s = 0.f;
#pragma unroll 4
    for (int n = 0; n < 128; ++n) s += __expf(dots[m * 128 + n] - mx);
    float l = mx + __logf(s);
    float inv = 1.f / s;
#pragma unroll 4
    for (int n = 0; n < 128; ++n)
      probs[m * 128 + n] = (_Float16)(__expf(dots[m * 128 + n] - mx) * inv);
    lse_s[(size_t)bh * kSN + c * kBucket + m] = l;
  }
  __syncthreads();

  // bo = probs (64x128) * v (128x64) : 16 tiles, 2 per wave
#pragma unroll
  for (int t = 0; t < 2; ++t) {
    int tile = wave * 2 + t;
    int mi = tile >> 2, ni = tile & 3;
    // batch all 4 k-step frag pairs, then the WMMA chain
    v16h af[4], bfr[4];
#pragma unroll
    for (int kk = 0; kk < 4; ++kk) {
      af[kk]  = load_a_frag(probs + mi * 16 * 128 + kk * 32, 128);
      bfr[kk] = load_b_frag(vT    + ni * 16 * 128 + kk * 32, 128);
    }
    v8f acc = {};
#pragma unroll
    for (int kk = 0; kk < 4; ++kk) {
      acc = __builtin_amdgcn_wmma_f32_16x16x32_f16(false, af[kk], false, bfr[kk],
                                                   (short)0, acc, false, false);
    }
    int mbase = mi * 16 + (lane >> 4) * 8;
    int e     = ni * 16 + (lane & 15);
#pragma unroll
    for (int r = 0; r < 8; ++r) {
      int m = mbase + r;
      bo[((size_t)bh * kSN + c * kBucket + m) * kE + e] = acc[r];
    }
  }
}

// ---------------- unsort + combine hash rounds via lse softmax ----------------
__global__ void combine_kernel(const float* __restrict__ bo, const float* __restrict__ lse_s,
                               const int* __restrict__ undo, _Float16* __restrict__ att16) {
  int idx = blockIdx.x * blockDim.x + threadIdx.x;
  if (idx >= kBH * kL) return;
  int bh = idx / kL, n = idx % kL;
  int k0 = undo[(size_t)bh * kSN + n];
  int k1 = undo[(size_t)bh * kSN + kL + n];
  float l0 = lse_s[(size_t)bh * kSN + k0];
  float l1 = lse_s[(size_t)bh * kSN + k1];
  float mx = fmaxf(l0, l1);
  float w0 = __expf(l0 - mx), w1 = __expf(l1 - mx);
  float inv = 1.f / (w0 + w1);
  w0 *= inv; w1 *= inv;
  const float* o0 = bo + ((size_t)bh * kSN + k0) * kE;
  const float* o1 = bo + ((size_t)bh * kSN + k1) * kE;
  int bb = bh / kH, h = bh % kH;
  _Float16* dst = att16 + (((size_t)bb * kL + n) * kH + h) * kE;
#pragma unroll
  for (int d = 0; d < kE; ++d) dst[d] = (_Float16)(w0 * o0[d] + w1 * o1[d]);
}

// ---------------- host-side launch ----------------
extern "C" void kernel_launch(void* const* d_in, const int* in_sizes, int n_in,
                              void* d_out, int out_size, void* d_ws, size_t ws_size,
                              hipStream_t stream) {
  const float* x      = (const float*)d_in[0];
  const unsigned char* pmask = (const unsigned char*)d_in[1];
  const float* w_qk   = (const float*)d_in[2];
  const float* w_v    = (const float*)d_in[3];
  const float* w_out  = (const float*)d_in[4];
  const float* rot    = (const float*)d_in[5];
  float* out = (float*)d_out;

  char* ws = (char*)d_ws;
  auto alloc = [&](size_t bytes) {
    char* p = ws;
    ws += (bytes + 255) & ~(size_t)255;
    return p;
  };
  const size_t M = (size_t)kB * kL;                       // 16384
  _Float16* xf16   = (_Float16*)alloc(M * kHid * 2);
  _Float16* wtqk   = (_Float16*)alloc((size_t)kHid * kHid * 2);
  _Float16* wtv    = (_Float16*)alloc((size_t)kHid * kHid * 2);
  _Float16* wtout  = (_Float16*)alloc((size_t)kHid * kHid * 2);
  float*    qkf    = (float*)alloc((size_t)kBH * kL * kE * 4);
  float*    vf     = (float*)alloc((size_t)kBH * kL * kE * 4);
  int*      bucks  = (int*)alloc((size_t)kBH * kSN * 4);
  int*      stick  = (int*)alloc((size_t)kBH * kSN * 4);
  int*      undo   = (int*)alloc((size_t)kBH * kSN * 4);
  int*      tposb  = (int*)alloc((size_t)kBH * kSN * 4);
  unsigned char* smaskb = (unsigned char*)alloc((size_t)kBH * kSN);
  _Float16* sq16   = (_Float16*)alloc((size_t)kBH * kSN * kE * 2);
  _Float16* sk16   = (_Float16*)alloc((size_t)kBH * kSN * kE * 2);
  _Float16* sv16   = (_Float16*)alloc((size_t)kBH * kSN * kE * 2);
  float*    bob    = (float*)alloc((size_t)kBH * kSN * kE * 4);
  float*    lseb   = (float*)alloc((size_t)kBH * kSN * 4);
  _Float16* att16  = (_Float16*)alloc(M * kHid * 2);

  // 1) convert x to f16; transpose+convert weights
  {
    int n = (int)(M * kHid);
    conv_f32_to_f16<<<(n + 255) / 256, 256, 0, stream>>>(x, xf16, n);
    int nw = kHid * kHid;
    conv_transpose_w<<<(nw + 255) / 256, 256, 0, stream>>>(w_qk, wtqk);
    conv_transpose_w<<<(nw + 255) / 256, 256, 0, stream>>>(w_v, wtv);
    conv_transpose_w<<<(nw + 255) / 256, 256, 0, stream>>>(w_out, wtout);
  }
  // 2) projections (scatter into [b,h,l,e] layout)
  {
    dim3 grid(kHid / 64, (unsigned)(M / 64));
    gemm_f16f32<<<grid, 128, 0, stream>>>(xf16, wtqk, qkf, (int)M, kHid, kHid, 1);
    gemm_f16f32<<<grid, 128, 0, stream>>>(xf16, wtv,  vf,  (int)M, kHid, kHid, 1);
  }
  // 3) hashing
  {
    int n = kBH * kL;
    hash_kernel<<<(n + 127) / 128, 128, 0, stream>>>(qkf, rot, bucks);
  }
  // 4) stable sort per BH
  sort_kernel<<<kBH, 128, 0, stream>>>(bucks, stick, undo);
  // 5) gather sorted q / keys / v
  {
    int n = kBH * kSN;
    gather_kernel<<<(n + 255) / 256, 256, 0, stream>>>(qkf, vf, pmask, stick,
                                                       sq16, sk16, sv16, tposb, smaskb);
  }
  // 6) chunked local attention (TDM-staged tiles)
  attn_kernel<<<dim3(kNCh, kBH), 256, 0, stream>>>(sq16, sk16, sv16, tposb, smaskb, bob, lseb);
  // 7) unsort + combine hash rounds
  {
    int n = kBH * kL;
    combine_kernel<<<(n + 255) / 256, 256, 0, stream>>>(bob, lseb, undo, att16);
  }
  // 8) output projection
  {
    dim3 grid(kHid / 64, (unsigned)(M / 64));
    gemm_f16f32<<<grid, 128, 0, stream>>>(att16, wtout, out, (int)M, kHid, kHid, 0);
  }
}
